// GUEnc_V_38465727103473
// MI455X (gfx1250) — compile-verified, matrix operand order checked by reading, not run on hardware
//
#include <hip/hip_runtime.h>
#include <hip/hip_bf16.h>
#include <stdint.h>

// Problem constants (from reference)
#define NN  50000
#define EE  800000
#define HH  128
#define EHH 16
#define N1  25000
#define N2  12500
#define N3  6250
#define BN_SCALE 0.9999950000374997f

typedef __attribute__((ext_vector_type(16))) _Float16 v16h;
typedef __attribute__((ext_vector_type(8)))  _Float16 v8h;
typedef __attribute__((ext_vector_type(8)))  float    v8f;

// ---------- helpers ----------
__device__ __forceinline__ unsigned f2key(float f) {
  union { float f; unsigned u; } c; c.f = f;
  return (c.u & 0x80000000u) ? ~c.u : (c.u | 0x80000000u);
}
__device__ __forceinline__ float key2f(unsigned u) {
  union { float f; unsigned u; } c;
  c.u = (u & 0x80000000u) ? (u ^ 0x80000000u) : ~u;
  return c.f;
}

// ---------- small init ----------
__global__ void k_init_small(unsigned* mm) {
  if (threadIdx.x == 0) { mm[0] = 0u; mm[1] = 0xFFFFFFFFu; }
}

// ---------- edge embedding + min/max ----------
__global__ void k_edge_embed(const float* __restrict__ ea, const float* __restrict__ Ww,
                             const float* __restrict__ Wb, float* __restrict__ ew,
                             unsigned* __restrict__ mm, int E_) {
  int e = blockIdx.x * blockDim.x + threadIdx.x;
  unsigned kmax = 0u, kmin = 0xFFFFFFFFu;
  if (e < E_) {
    float s = Wb[0];
#pragma unroll
    for (int j = 0; j < EHH; ++j) s += ea[(size_t)e * EHH + j] * Ww[j];
    ew[e] = s;
    unsigned key = f2key(s);
    kmax = key; kmin = key;
  }
  for (int off = 16; off; off >>= 1) {
    unsigned a = (unsigned)__shfl_down((int)kmax, off);
    unsigned b = (unsigned)__shfl_down((int)kmin, off);
    kmax = (a > kmax) ? a : kmax;
    kmin = (b < kmin) ? b : kmin;
  }
  if ((threadIdx.x & 31) == 0) { atomicMax(&mm[0], kmax); atomicMin(&mm[1], kmin); }
}

__global__ void k_ew_norm(float* __restrict__ ew, const unsigned* __restrict__ mm, int E_) {
  int e = blockIdx.x * blockDim.x + threadIdx.x;
  if (e >= E_) return;
  float mx = key2f(mm[0]), mn = key2f(mm[1]);
  ew[e] = (ew[e] - mn) / ((mx - mn) + 1e-7f);
}

// ---------- elementwise ----------
__global__ void k_bn(const float* __restrict__ x, const float* __restrict__ g,
                     const float* __restrict__ b, float* __restrict__ y, int total) {
  int i = blockIdx.x * blockDim.x + threadIdx.x;
  if (i >= total) return;
  int c = i & (HH - 1);
  y[i] = x[i] * (g[c] * BN_SCALE) + b[c];
}
__global__ void k_lrelu(float* __restrict__ x, int total) {
  int i = blockIdx.x * blockDim.x + threadIdx.x;
  if (i >= total) return;
  float v = x[i];
  x[i] = v > 0.f ? v : 0.01f * v;
}
__global__ void k_fillf(float* __restrict__ p, float v, int n) {
  int i = blockIdx.x * blockDim.x + threadIdx.x;
  if (i < n) p[i] = v;
}

// ---------- weight pre-convert: W[k][n] f32 -> Wh[n][k] f16 (transposed) ----------
__global__ void k_convW(const float* __restrict__ W, _Float16* __restrict__ Wh) {
  int i = blockIdx.x * blockDim.x + threadIdx.x;
  if (i >= HH * HH) return;
  int k = i >> 7, n = i & (HH - 1);
  Wh[(size_t)n * HH + k] = (_Float16)W[i];
}

// ---------- WMMA GEMM: C[n x 128] = A[n x 128] * W[128 x 128]
// One wave computes a 16x128 strip: 8 f32 accumulators, 32 v_wmma per wave.
// A: f32 row-major, loaded as float4 and converted. B: pre-converted f16,
// transposed layout Wh[n][k] so each lane's fragment is two 16B vector loads.
__global__ void k_gemm_wmma(const float* __restrict__ A, const _Float16* __restrict__ Wh,
                            float* __restrict__ C, int n) {
  const int lane = threadIdx.x & 31;
  const int half = lane >> 4;     // 0/1
  const int l16  = lane & 15;
  const int m0   = blockIdx.x << 4;
  int row = m0 + l16;
  if (row >= n) row = n - 1;      // clamp: out-of-range rows computed but never stored
  const float* arow = A + (size_t)row * HH;
  v8f acc[8] = {};
#pragma unroll
  for (int k0 = 0; k0 < HH; k0 += 32) {
    const int kb = k0 + 8 * half; // lanes 0-15: K kb..kb+7 & kb+16..kb+23 (kb=k0); lanes 16-31: +8
    const float4 a0 = *(const float4*)(arow + kb);
    const float4 a1 = *(const float4*)(arow + kb + 4);
    const float4 a2 = *(const float4*)(arow + kb + 16);
    const float4 a3 = *(const float4*)(arow + kb + 20);
    v16h a;
    a[0] = (_Float16)a0.x;  a[1] = (_Float16)a0.y;  a[2] = (_Float16)a0.z;  a[3] = (_Float16)a0.w;
    a[4] = (_Float16)a1.x;  a[5] = (_Float16)a1.y;  a[6] = (_Float16)a1.z;  a[7] = (_Float16)a1.w;
    a[8] = (_Float16)a2.x;  a[9] = (_Float16)a2.y;  a[10] = (_Float16)a2.z; a[11] = (_Float16)a2.w;
    a[12] = (_Float16)a3.x; a[13] = (_Float16)a3.y; a[14] = (_Float16)a3.z; a[15] = (_Float16)a3.w;
#pragma unroll
    for (int t8 = 0; t8 < 8; ++t8) {
      const _Float16* wp = Wh + (size_t)(t8 * 16 + l16) * HH + kb;
      v8h blo = *(const v8h*)(wp);
      v8h bhi = *(const v8h*)(wp + 16);
      v16h b = __builtin_shufflevector(blo, bhi, 0, 1, 2, 3, 4, 5, 6, 7,
                                                 8, 9, 10, 11, 12, 13, 14, 15);
      acc[t8] = __builtin_amdgcn_wmma_f32_16x16x32_f16(false, a, false, b, (short)0,
                                                       acc[t8], false, false);
    }
  }
  // C/D layout: VGPR r -> M=r (lanes 0-15), M=r+8 (lanes 16-31); N = lane%16
#pragma unroll
  for (int r = 0; r < 8; ++r) {
    int m = m0 + r + 8 * half;
    if (m < n) {
      float* cp = C + (size_t)m * HH + l16;
#pragma unroll
      for (int t8 = 0; t8 < 8; ++t8) cp[t8 * 16] = acc[t8][r];
    }
  }
}

// ---------- GCN pieces ----------
__global__ void k_deg_acc(const int* __restrict__ dst, const float* __restrict__ ew,
                          float* __restrict__ deg, int E_) {
  int e = blockIdx.x * blockDim.x + threadIdx.x;
  if (e >= E_) return;
  atomicAdd(&deg[dst[e]], ew[e]);
}
__global__ void k_dinv(const float* __restrict__ deg, float* __restrict__ dinv,
                       float* __restrict__ selfco, float fill, int n) {
  int i = blockIdx.x * blockDim.x + threadIdx.x;
  if (i >= n) return;
  float d = deg[i];
  dinv[i] = rsqrtf(d);
  selfco[i] = fill / d;
}
__global__ void k_selfinit(const float* __restrict__ h, const float* __restrict__ selfco,
                           const float* __restrict__ b, float* __restrict__ out, int total) {
  int i = blockIdx.x * blockDim.x + threadIdx.x;
  if (i >= total) return;
  out[i] = selfco[i >> 7] * h[i] + b[i & (HH - 1)];
}
// one wave per edge; float4 per lane; L2-resident atomics
__global__ void k_scatter(const float* __restrict__ h, const int* __restrict__ src,
                          const int* __restrict__ dst, const float* __restrict__ ew,
                          const float* __restrict__ dinv, float* __restrict__ out, int E_) {
  int e = blockIdx.x * (blockDim.x >> 5) + (threadIdx.x >> 5);
  if (e >= E_) return;
  int s = src[e], d = dst[e];
  float c = dinv[s] * ew[e] * dinv[d];
  if (c == 0.f) return; // masked edges contribute nothing
  int lane = threadIdx.x & 31;
  const float4 v = ((const float4*)(h + (size_t)s * HH))[lane];
  float* op = out + (size_t)d * HH + lane * 4;
  atomicAdd(op + 0, c * v.x);
  atomicAdd(op + 1, c * v.y);
  atomicAdd(op + 2, c * v.z);
  atomicAdd(op + 3, c * v.w);
}

// ---------- top-k pooling ----------
__global__ void k_pnorm(const float* __restrict__ p, float* __restrict__ inv) {
  __shared__ float sh[HH];
  int t = threadIdx.x;
  float v = p[t];
  sh[t] = v * v;
  __syncthreads();
  for (int s = HH / 2; s; s >>= 1) { if (t < s) sh[t] += sh[t + s]; __syncthreads(); }
  if (t == 0) inv[0] = rsqrtf(sh[0]);
}
__global__ void k_score(const float* __restrict__ x, const float* __restrict__ p,
                        const float* __restrict__ inv, float* __restrict__ score, int n) {
  int i = blockIdx.x * (blockDim.x >> 5) + (threadIdx.x >> 5);
  if (i >= n) return;
  int lane = threadIdx.x & 31;
  float4 xv = ((const float4*)(x + (size_t)i * HH))[lane];
  float4 pv = ((const float4*)p)[lane];
  float d = xv.x * pv.x + xv.y * pv.y + xv.z * pv.z + xv.w * pv.w;
  for (int off = 16; off; off >>= 1) d += __shfl_down(d, off);
  if (lane == 0) score[i] = tanhf(d * inv[0]);
}
// exact k-th largest via 4-pass byte radix select; single workgroup
__global__ void k_radix_select(const float* __restrict__ score, int n, int k,
                               unsigned* __restrict__ sel) {
  __shared__ unsigned hist[256];
  __shared__ unsigned sprefix, sremain;
  int tid = threadIdx.x;
  if (tid == 0) { sprefix = 0u; sremain = (unsigned)k; }
  __syncthreads();
  for (int d = 3; d >= 0; --d) {
    if (tid < 256) hist[tid] = 0u;
    __syncthreads();
    unsigned prefix = sprefix;
    unsigned mask_hi = (d == 3) ? 0u : (0xFFFFFFFFu << ((d + 1) * 8));
    for (int i = tid; i < n; i += blockDim.x) {
      unsigned u = f2key(score[i]);
      if ((u & mask_hi) == prefix) atomicAdd(&hist[(u >> (d * 8)) & 0xFFu], 1u);
    }
    __syncthreads();
    if (tid == 0) {
      unsigned rem = sremain, cum = 0u, v = 0u;
      for (int b = 255; b >= 0; --b) {
        if (cum + hist[b] >= rem) { v = (unsigned)b; break; }
        cum += hist[b];
      }
      sprefix = prefix | (v << (d * 8));
      sremain = rem - cum;
    }
    __syncthreads();
  }
  if (tid == 0) {
    sel[0] = sprefix;                 // key of k-th largest
    sel[1] = (unsigned)k - sremain;   // count strictly greater
    sel[2] = sremain;                 // how many equal-to-T to take
    sel[3] = 0u;                      // gt counter
    sel[4] = 0u;                      // eq counter
  }
}
__global__ void k_topk_build(const float* __restrict__ score, int n, int k,
                             unsigned* sel, int* __restrict__ pos,
                             int* __restrict__ perm, float* __restrict__ vals) {
  int i = blockIdx.x * blockDim.x + threadIdx.x;
  if (i >= n) return;
  unsigned u = f2key(score[i]);
  unsigned T = sel[0], cgt = sel[1], etake = sel[2];
  int idx = -1;
  if (u > T) {
    idx = (int)atomicAdd(&sel[3], 1u);
  } else if (u == T) {
    unsigned t = atomicAdd(&sel[4], 1u);
    if (t < etake) idx = (int)(cgt + t);
  }
  if (idx >= k) idx = -1; // safety
  pos[i] = idx;
  if (idx >= 0) { perm[idx] = i; vals[idx] = score[i]; }
}
__global__ void k_pool_gather(const float* __restrict__ x, const int* __restrict__ perm,
                              const float* __restrict__ vals, float* __restrict__ xp,
                              int k, int n_in) {
  int i = blockIdx.x * blockDim.x + threadIdx.x;
  if (i >= k * HH) return;
  int j = i >> 7, c = i & (HH - 1);
  int s = perm[j];
  s = s < 0 ? 0 : (s >= n_in ? n_in - 1 : s);
  xp[i] = x[(size_t)s * HH + c] * vals[j];
}
__global__ void k_edge_remap(const int* __restrict__ src, const int* __restrict__ dst,
                             const float* __restrict__ ew, const int* __restrict__ pos,
                             int* __restrict__ nsrc, int* __restrict__ ndst,
                             float* __restrict__ new_ew, int E_) {
  int e = blockIdx.x * blockDim.x + threadIdx.x;
  if (e >= E_) return;
  int ns = pos[src[e]], nd = pos[dst[e]];
  bool ok = (ns >= 0) && (nd >= 0);
  nsrc[e] = ok ? ns : 0;
  ndst[e] = ok ? nd : 0;
  new_ew[e] = ok ? ew[e] : 0.f;
}
__global__ void k_unpool_add(float* __restrict__ res, const int* __restrict__ perm,
                             const float* __restrict__ h, int k, int n_out) {
  int i = blockIdx.x * blockDim.x + threadIdx.x;
  if (i >= k * HH) return;
  int j = i >> 7, c = i & (HH - 1);
  int r = perm[j];
  r = r < 0 ? 0 : (r >= n_out ? n_out - 1 : r);
  res[(size_t)r * HH + c] += h[i];
}

// ---------- readout ----------
__global__ void k_sumpool(const float* __restrict__ h, float* __restrict__ g, int n) {
  int c = threadIdx.x;             // 0..127
  int r0 = blockIdx.x * 128;
  int rend = r0 + 128; if (rend > n) rend = n;
  float s = 0.f;
  for (int r = r0; r < rend; ++r) s += h[(size_t)r * HH + c];
  atomicAdd(&g[c], s);
}
__global__ void k_final(const float* __restrict__ g, const float* __restrict__ Wr,
                        const float* __restrict__ br, const float* __restrict__ gr,
                        const float* __restrict__ brn, float* __restrict__ out) {
  __shared__ float sg[HH];
  int j = threadIdx.x;
  sg[j] = g[j];
  __syncthreads();
  float acc = br[j];
#pragma unroll 4
  for (int i = 0; i < HH; ++i) acc += sg[i] * Wr[(size_t)i * HH + j];
  out[j] = acc * (gr[j] * BN_SCALE) + brn[j];
}

// ---------- host ----------
extern "C" void kernel_launch(void* const* d_in, const int* in_sizes, int n_in_cnt,
                              void* d_out, int out_size, void* d_ws, size_t ws_size,
                              hipStream_t stream) {
  const float* x      = (const float*)d_in[0];
  const int*   eidx   = (const int*)d_in[1];
  const float* eattr  = (const float*)d_in[2];
  const float* We_w   = (const float*)d_in[3];
  const float* We_b   = (const float*)d_in[4];
  const float* Wdown  = (const float*)d_in[5];
  const float* bdown  = (const float*)d_in[6];
  const float* Wpool  = (const float*)d_in[7];
  const float* Wup    = (const float*)d_in[8];
  const float* bup    = (const float*)d_in[9];
  const float* gnorm  = (const float*)d_in[10];
  const float* bnorm  = (const float*)d_in[11];
  const float* Wr     = (const float*)d_in[12];
  const float* br     = (const float*)d_in[13];
  const float* gr     = (const float*)d_in[14];
  const float* brn    = (const float*)d_in[15];

  const int* src0 = eidx;
  const int* dst0 = eidx + EE;

  // ---- bump allocator over d_ws (needs ~160 MB) ----
  char* wsb = (char*)d_ws;
  size_t off = 0;
  auto alloc = [&](size_t bytes) -> void* {
    off = (off + 255) & ~(size_t)255;
    void* p = wsb + off;
    off += bytes;
    return p;
  };
  float* ew0 = (float*)alloc(EE * 4);
  int*   s1  = (int*)alloc(EE * 4);  int* d1 = (int*)alloc(EE * 4);  float* w1 = (float*)alloc(EE * 4);
  int*   s2  = (int*)alloc(EE * 4);  int* d2 = (int*)alloc(EE * 4);  float* w2 = (float*)alloc(EE * 4);
  int*   s3  = (int*)alloc(EE * 4);  int* d3 = (int*)alloc(EE * 4);  float* w3 = (float*)alloc(EE * 4);
  float* xs0 = (float*)alloc((size_t)NN * HH * 4);
  float* xs1 = (float*)alloc((size_t)N1 * HH * 4);
  float* xs2 = (float*)alloc((size_t)N2 * HH * 4);
  float* bufA = (float*)alloc((size_t)NN * HH * 4);
  float* bufB = (float*)alloc((size_t)NN * HH * 4);
  float* bufC = (float*)alloc((size_t)NN * HH * 4);
  float* deg    = (float*)alloc(NN * 4);
  float* dinv   = (float*)alloc(NN * 4);
  float* selfco = (float*)alloc(NN * 4);
  float* score  = (float*)alloc(NN * 4);
  float* vals   = (float*)alloc(NN * 4);
  int* pos   = (int*)alloc(NN * 4);
  int* perm1 = (int*)alloc(N1 * 4);
  int* perm2 = (int*)alloc(N2 * 4);
  int* perm3 = (int*)alloc(N3 * 4);
  float* gsum = (float*)alloc(HH * 4);
  _Float16* Wh = (_Float16*)alloc((size_t)HH * HH * 2); // transposed f16 weights
  unsigned* mm  = (unsigned*)alloc(8 * 4);   // [0]=max key, [1]=min key
  unsigned* sel = (unsigned*)alloc(8 * 4);   // T, cnt_gt, eq_take, gt_ctr, eq_ctr
  float* pnorm_inv = (float*)alloc(4);
  (void)ws_size; (void)in_sizes; (void)n_in_cnt; (void)out_size;

  auto gcn = [&](const float* xin, const int* src, const int* dst, const float* ew,
                 const float* Wt, const float* bias, float fillv, int n,
                 float* hbuf, float* obuf) {
    k_convW<<<(HH * HH + 255) / 256, 256, 0, stream>>>(Wt, Wh);
    k_gemm_wmma<<<(n + 15) / 16, 32, 0, stream>>>(xin, Wh, hbuf, n);
    k_fillf<<<(n + 255) / 256, 256, 0, stream>>>(deg, fillv, n);
    k_deg_acc<<<(EE + 255) / 256, 256, 0, stream>>>(dst, ew, deg, EE);
    k_dinv<<<(n + 255) / 256, 256, 0, stream>>>(deg, dinv, selfco, fillv, n);
    k_selfinit<<<((n * HH) + 255) / 256, 256, 0, stream>>>(hbuf, selfco, bias, obuf, n * HH);
    k_scatter<<<(EE + 7) / 8, 256, 0, stream>>>(hbuf, src, dst, ew, dinv, obuf, EE);
  };
  auto pool = [&](const float* hin, const int* src, const int* dst, const float* ew,
                  const float* pvec, int n_cur, int k,
                  float* hout, int* nsrc, int* ndst, float* new_ew, int* permv) {
    k_pnorm<<<1, HH, 0, stream>>>(pvec, pnorm_inv);
    k_score<<<(n_cur + 7) / 8, 256, 0, stream>>>(hin, pvec, pnorm_inv, score, n_cur);
    k_radix_select<<<1, 1024, 0, stream>>>(score, n_cur, k, sel);
    k_topk_build<<<(n_cur + 255) / 256, 256, 0, stream>>>(score, n_cur, k, sel, pos, permv, vals);
    k_pool_gather<<<((k * HH) + 255) / 256, 256, 0, stream>>>(hin, permv, vals, hout, k, n_cur);
    k_edge_remap<<<(EE + 255) / 256, 256, 0, stream>>>(src, dst, ew, pos, nsrc, ndst, new_ew, EE);
  };

  // ---- edge embedding + min/max normalization ----
  k_init_small<<<1, 32, 0, stream>>>(mm);
  k_edge_embed<<<(EE + 255) / 256, 256, 0, stream>>>(eattr, We_w, We_b, ew0, mm, EE);
  k_ew_norm<<<(EE + 255) / 256, 256, 0, stream>>>(ew0, mm, EE);

  // ---- level 0: bn -> gcn -> lrelu ----
  k_bn<<<((NN * HH) + 255) / 256, 256, 0, stream>>>(x, gnorm, bnorm, bufB, NN * HH);
  gcn(bufB, src0, dst0, ew0, Wdown, bdown, 1.0f, NN, bufC, bufA);
  k_lrelu<<<((NN * HH) + 255) / 256, 256, 0, stream>>>(bufA, NN * HH);
  hipMemcpyAsync(xs0, bufA, (size_t)NN * HH * 4, hipMemcpyDeviceToDevice, stream);

  // ---- down i=0: 50000 -> 25000 ----
  pool(bufA, src0, dst0, ew0, Wpool + 0 * HH, NN, N1, bufB, s1, d1, w1, perm1);
  k_bn<<<((N1 * HH) + 255) / 256, 256, 0, stream>>>(bufB, gnorm + 1 * HH, bnorm + 1 * HH, bufB, N1 * HH);
  gcn(bufB, s1, d1, w1, Wdown + 1 * HH * HH, bdown + 1 * HH, 1.0f, N1, bufC, bufA);
  k_lrelu<<<((N1 * HH) + 255) / 256, 256, 0, stream>>>(bufA, N1 * HH);
  hipMemcpyAsync(xs1, bufA, (size_t)N1 * HH * 4, hipMemcpyDeviceToDevice, stream);

  // ---- down i=1: 25000 -> 12500 ----
  pool(bufA, s1, d1, w1, Wpool + 1 * HH, N1, N2, bufB, s2, d2, w2, perm2);
  k_bn<<<((N2 * HH) + 255) / 256, 256, 0, stream>>>(bufB, gnorm + 2 * HH, bnorm + 2 * HH, bufB, N2 * HH);
  gcn(bufB, s2, d2, w2, Wdown + 2 * HH * HH, bdown + 2 * HH, 1.0f, N2, bufC, bufA);
  k_lrelu<<<((N2 * HH) + 255) / 256, 256, 0, stream>>>(bufA, N2 * HH);
  hipMemcpyAsync(xs2, bufA, (size_t)N2 * HH * 4, hipMemcpyDeviceToDevice, stream);

  // ---- down i=2: 12500 -> 6250 (no lrelu/save) ----
  pool(bufA, s2, d2, w2, Wpool + 2 * HH, N2, N3, bufB, s3, d3, w3, perm3);
  k_bn<<<((N3 * HH) + 255) / 256, 256, 0, stream>>>(bufB, gnorm + 3 * HH, bnorm + 3 * HH, bufB, N3 * HH);
  gcn(bufB, s3, d3, w3, Wdown + 3 * HH * HH, bdown + 3 * HH, 1.0f, N3, bufC, bufA);

  // ---- up i=0 (j=2): unpool to 12500 ----
  hipMemcpyAsync(bufB, xs2, (size_t)N2 * HH * 4, hipMemcpyDeviceToDevice, stream);
  k_unpool_add<<<((N3 * HH) + 255) / 256, 256, 0, stream>>>(bufB, perm3, bufA, N3, N2);
  gcn(bufB, s2, d2, w2, Wup + 0 * HH * HH, bup + 0 * HH, 2.0f, N2, bufC, bufA);
  k_lrelu<<<((N2 * HH) + 255) / 256, 256, 0, stream>>>(bufA, N2 * HH);

  // ---- up i=1 (j=1): unpool to 25000 ----
  hipMemcpyAsync(bufB, xs1, (size_t)N1 * HH * 4, hipMemcpyDeviceToDevice, stream);
  k_unpool_add<<<((N2 * HH) + 255) / 256, 256, 0, stream>>>(bufB, perm2, bufA, N2, N1);
  gcn(bufB, s1, d1, w1, Wup + 1 * HH * HH, bup + 1 * HH, 2.0f, N1, bufC, bufA);
  k_lrelu<<<((N1 * HH) + 255) / 256, 256, 0, stream>>>(bufA, N1 * HH);

  // ---- up i=2 (j=0): unpool to 50000 (no lrelu) ----
  hipMemcpyAsync(bufB, xs0, (size_t)NN * HH * 4, hipMemcpyDeviceToDevice, stream);
  k_unpool_add<<<((N1 * HH) + 255) / 256, 256, 0, stream>>>(bufB, perm1, bufA, N1, NN);
  gcn(bufB, src0, dst0, ew0, Wup + 2 * HH * HH, bup + 2 * HH, 2.0f, NN, bufC, bufA);

  // ---- readout ----
  k_fillf<<<1, HH, 0, stream>>>(gsum, 0.f, HH);
  k_sumpool<<<(NN + 127) / 128, 128, 0, stream>>>(bufA, gsum, NN);
  k_final<<<1, HH, 0, stream>>>(gsum, Wr, br, gr, brn, (float*)d_out);
}
